// AttnDecoder_47278999994969
// MI455X (gfx1250) — compile-verified, matrix operand order checked by reading, not run on hardware
//
#include <hip/hip_runtime.h>
#include <hip/hip_bf16.h>

// ---------------------------------------------------------------------------
// AttnDecoder single step, CDNA5 (gfx1250, wave32).
// All dense layers run on v_wmma_f32_16x16x32_f16 (f16 in, f32 accumulate).
// The 2 GB encoder_outputs stream (context einsum) is the roofline bottleneck
// (~86 us @ 23.3 TB/s); it is done with coalesced float4 streaming + prefetch.
// ---------------------------------------------------------------------------

#define B_   1024
#define H_   1024
#define T_   512
#define NUM_ 44
#define IN_  128
#define AK_  (IN_ + 2 * H_)   // 2176  (attention K)
#define CK_  (IN_ + H_)       // 1152  (combine K)
#define G_   (4 * H_)         // 4096  (gates N)

typedef __attribute__((ext_vector_type(16))) _Float16 v16h;
typedef __attribute__((ext_vector_type(8)))  _Float16 v8h;
typedef __attribute__((ext_vector_type(8)))  float    v8f;

// ---- fp32 -> f16 repack ----------------------------------------------------
__global__ void f32_to_f16_kernel(const float* __restrict__ src,
                                  _Float16* __restrict__ dst, int n) {
    int i = blockIdx.x * blockDim.x + threadIdx.x;
    if (i < n) dst[i] = (_Float16)src[i];
}

// ---- K-major 16x32 f16 fragment load (identical layout for A and B^T) ------
// lane l holds row (l&15); halfs at K = ktile + (l>>4)*8 + {0..7} and +16+{0..7}
__device__ __forceinline__ v16h load_frag_kmajor(const _Float16* __restrict__ base,
                                                 int row, int ldk, int ktile, int lane) {
    const int r  = row + (lane & 15);
    const int kb = ktile + ((lane >> 4) << 3);
    const _Float16* p = base + (size_t)r * ldk + kb;
    v8h lo = *(const v8h*)(p);
    v8h hi = *(const v8h*)(p + 16);
    v16h f;
#pragma unroll
    for (int i = 0; i < 8; ++i) { f[i] = lo[i]; f[8 + i] = hi[i]; }
    return f;
}

// ---- generic WMMA GEMM: C = A1 @ W1^T (+ A2 @ W2^T) + bias1 (+ bias2) ------
// A*: M x K (f16, row-major/K-major).  W*: N x K (f16, K-major).
// One 16x16 tile per wave; 8 waves per 256-thread block.
__global__ void wmma_gemm_dual(const _Float16* __restrict__ A1,
                               const _Float16* __restrict__ W1, int K1,
                               const _Float16* __restrict__ A2,
                               const _Float16* __restrict__ W2, int K2,
                               const float* __restrict__ bias1,
                               const float* __restrict__ bias2,
                               float* __restrict__ Cf,
                               _Float16* __restrict__ Ch,
                               int M, int N) {
    const int lane   = threadIdx.x & 31;
    const int wave   = blockIdx.x * (blockDim.x >> 5) + (threadIdx.x >> 5);
    const int ntiles = N >> 4;
    const int tm     = (wave / ntiles) << 4;
    const int tn     = (wave % ntiles) << 4;
    if (tm >= M) return;

    v8f acc = {};
    for (int k = 0; k < K1; k += 32) {
        v16h a = load_frag_kmajor(A1, tm, K1, k, lane);
        v16h w = load_frag_kmajor(W1, tn, K1, k, lane);
        acc = __builtin_amdgcn_wmma_f32_16x16x32_f16(false, a, false, w,
                                                     (short)0, acc, false, false);
    }
    if (A2) {
        for (int k = 0; k < K2; k += 32) {
            v16h a = load_frag_kmajor(A2, tm, K2, k, lane);
            v16h w = load_frag_kmajor(W2, tn, K2, k, lane);
            acc = __builtin_amdgcn_wmma_f32_16x16x32_f16(false, a, false, w,
                                                         (short)0, acc, false, false);
        }
    }

    const int n  = tn + (lane & 15);
    float bias = 0.0f;
    if (bias1) bias += bias1[n];
    if (bias2) bias += bias2[n];
    const int m0 = tm + ((lane >> 4) << 3);
#pragma unroll
    for (int r = 0; r < 8; ++r) {
        float v = acc[r] + bias;
        size_t idx = (size_t)(m0 + r) * N + n;
        if (Cf) Cf[idx] = v;
        if (Ch) Ch[idx] = (_Float16)v;
    }
}

// ---- embeddings + concat: x_rnn (f32), attn_in row (f16), comb_in[0:128] ---
__global__ void build_inputs_kernel(const float* __restrict__ x,
                                    const int* __restrict__ x_emb,
                                    const float* __restrict__ h0,
                                    const float* __restrict__ c0,
                                    const float* __restrict__ emb0,
                                    const float* __restrict__ emb1,
                                    const float* __restrict__ emb2,
                                    const float* __restrict__ emb3,
                                    float* __restrict__ x_rnn,
                                    _Float16* __restrict__ attn_in,
                                    _Float16* __restrict__ comb_in) {
    const int b   = blockIdx.x;
    const int tid = threadIdx.x;   // 256
    if (tid < IN_) {
        float v;
        if (tid < 44)       v = x[b * NUM_ + tid];
        else if (tid < 108) v = emb0[x_emb[b * 4 + 0] * 64 + (tid - 44)];
        else if (tid < 116) v = emb1[x_emb[b * 4 + 1] * 8  + (tid - 108)];
        else if (tid < 124) v = emb2[x_emb[b * 4 + 2] * 8  + (tid - 116)];
        else                v = emb3[x_emb[b * 4 + 3] * 4  + (tid - 124)];
        x_rnn[b * IN_ + tid] = v;
        attn_in[(size_t)b * AK_ + tid] = (_Float16)v;
        comb_in[(size_t)b * CK_ + tid] = (_Float16)v;
    }
    for (int j = IN_ + tid; j < AK_; j += 256) {
        float v = (j < IN_ + H_) ? h0[(size_t)b * H_ + (j - IN_)]
                                 : c0[(size_t)b * H_ + (j - IN_ - H_)];
        attn_in[(size_t)b * AK_ + j] = (_Float16)v;
    }
}

// ---- row softmax over T=512 ------------------------------------------------
__global__ void softmax_kernel(const float* __restrict__ logits,
                               float* __restrict__ aw) {
    const int b   = blockIdx.x;
    const int tid = threadIdx.x;   // 256
    __shared__ float red[256];
    float v0 = logits[(size_t)b * T_ + tid];
    float v1 = logits[(size_t)b * T_ + 256 + tid];
    red[tid] = fmaxf(v0, v1);
    __syncthreads();
    for (int s = 128; s > 0; s >>= 1) {
        if (tid < s) red[tid] = fmaxf(red[tid], red[tid + s]);
        __syncthreads();
    }
    float m = red[0];
    __syncthreads();
    float e0 = __expf(v0 - m), e1 = __expf(v1 - m);
    red[tid] = e0 + e1;
    __syncthreads();
    for (int s = 128; s > 0; s >>= 1) {
        if (tid < s) red[tid] += red[tid + s];
        __syncthreads();
    }
    float inv = 1.0f / red[0];
    aw[(size_t)b * T_ + tid]       = e0 * inv;
    aw[(size_t)b * T_ + 256 + tid] = e1 * inv;
}

// ---- ctx[b,h] = sum_t aw[b,t] * enc[t,b,h]  (the 2 GB stream) --------------
// one block per b; 256 threads x float4 = full H row; coalesced 4 KB per t.
__global__ void ctx_kernel(const float* __restrict__ aw,
                           const float* __restrict__ enc,
                           _Float16* __restrict__ comb_in) {
    const int b   = blockIdx.x;
    const int tid = threadIdx.x;   // 256
    const int h   = tid * 4;
    __shared__ float saw[T_];
    saw[tid]       = aw[(size_t)b * T_ + tid];
    saw[tid + 256] = aw[(size_t)b * T_ + 256 + tid];
    __syncthreads();
    float4 acc = make_float4(0.f, 0.f, 0.f, 0.f);
    for (int t = 0; t < T_; ++t) {
        if (t + 16 < T_)
            __builtin_prefetch(enc + ((size_t)(t + 16) * B_ + b) * H_ + h, 0, 0);
        const float  s = saw[t];
        const float4 e = *(const float4*)(enc + ((size_t)t * B_ + b) * H_ + h);
        acc.x += s * e.x; acc.y += s * e.y; acc.z += s * e.z; acc.w += s * e.w;
    }
    _Float16* dst = comb_in + (size_t)b * CK_ + IN_ + h;
    dst[0] = (_Float16)acc.x; dst[1] = (_Float16)acc.y;
    dst[2] = (_Float16)acc.z; dst[3] = (_Float16)acc.w;
}

// ---- LSTM cell elementwise -------------------------------------------------
__device__ __forceinline__ float sig_(float x) { return 1.0f / (1.0f + __expf(-x)); }

__global__ void lstm_elem_kernel(const float* __restrict__ gates,
                                 const float* __restrict__ c_prev,
                                 float* __restrict__ h_out,
                                 float* __restrict__ c_out,
                                 _Float16* __restrict__ h_h) {
    int idx = blockIdx.x * blockDim.x + threadIdx.x;    // B*H
    int b = idx >> 10, h = idx & (H_ - 1);
    const float* g = gates + (size_t)b * G_;
    float i  = sig_(g[h]);
    float f  = sig_(g[H_ + h]);
    float gg = tanhf(g[2 * H_ + h]);
    float o  = sig_(g[3 * H_ + h]);
    float c  = f * c_prev[idx] + i * gg;
    float hh = o * tanhf(c);
    c_out[idx] = c;
    h_out[idx] = hh;
    if (h_h) h_h[idx] = (_Float16)hh;
}

// ---- out[b] = dot(h2[b,:], pred_w) + pred_b --------------------------------
__global__ void pred_kernel(const float* __restrict__ h2,
                            const float* __restrict__ pw,
                            const float* __restrict__ pb,
                            float* __restrict__ out) {
    const int b   = blockIdx.x;
    const int tid = threadIdx.x;   // 256
    __shared__ float red[256];
    float s = 0.f;
    for (int h = tid; h < H_; h += 256) s += h2[(size_t)b * H_ + h] * pw[h];
    red[tid] = s;
    __syncthreads();
    for (int st = 128; st > 0; st >>= 1) {
        if (tid < st) red[tid] += red[tid + st];
        __syncthreads();
    }
    if (tid == 0) out[b] = red[0] + pb[0];
}

// ---------------------------------------------------------------------------
extern "C" void kernel_launch(void* const* d_in, const int* in_sizes, int n_in,
                              void* d_out, int out_size, void* d_ws, size_t ws_size,
                              hipStream_t stream) {
    (void)in_sizes; (void)n_in; (void)out_size; (void)ws_size;

    const float* x      = (const float*)d_in[0];
    const int*   x_emb  = (const int*)  d_in[1];
    const float* h0     = (const float*)d_in[2];   // (2,B,H)
    const float* c0     = (const float*)d_in[3];   // (2,B,H)
    const float* enc    = (const float*)d_in[4];   // (T,B,H)
    const float* emb0   = (const float*)d_in[5];
    const float* emb1   = (const float*)d_in[6];
    const float* emb2   = (const float*)d_in[7];
    const float* emb3   = (const float*)d_in[8];
    const float* attn_w = (const float*)d_in[9];   // (T, AK)
    const float* attn_b = (const float*)d_in[10];
    const float* comb_w = (const float*)d_in[11];  // (IN, CK)
    const float* comb_b = (const float*)d_in[12];
    const float* w_ih0  = (const float*)d_in[13];  // (G, IN)
    const float* w_hh0  = (const float*)d_in[14];  // (G, H)
    const float* b_ih0  = (const float*)d_in[15];
    const float* b_hh0  = (const float*)d_in[16];
    const float* w_ih1  = (const float*)d_in[17];  // (G, H)
    const float* w_hh1  = (const float*)d_in[18];  // (G, H)
    const float* b_ih1  = (const float*)d_in[19];
    const float* b_hh1  = (const float*)d_in[20];
    const float* pred_w = (const float*)d_in[21];
    const float* pred_b = (const float*)d_in[22];

    float* out = (float*)d_out;
    // d_out flat layout: out | h1 | h2 | c1 | c2 | aw
    const size_t OFF_H1 = 1024;
    const size_t OFF_H2 = OFF_H1 + (size_t)B_ * H_;
    const size_t OFF_C1 = OFF_H2 + (size_t)B_ * H_;
    const size_t OFF_C2 = OFF_C1 + (size_t)B_ * H_;
    const size_t OFF_AW = OFF_C2 + (size_t)B_ * H_;

    // ---- workspace carve-out (all 256 B aligned) ----
    char* ws = (char*)d_ws;
    size_t off = 0;
    auto alloc = [&](size_t bytes) -> char* {
        char* p = ws + off;
        off = (off + bytes + 255) & ~(size_t)255;
        return p;
    };
    _Float16* attn_in_h = (_Float16*)alloc((size_t)B_ * AK_ * 2);
    _Float16* attn_w_h  = (_Float16*)alloc((size_t)T_ * AK_ * 2);
    _Float16* comb_w_h  = (_Float16*)alloc((size_t)IN_ * CK_ * 2);
    _Float16* w_ih0_h   = (_Float16*)alloc((size_t)G_ * IN_ * 2);
    _Float16* w_hh0_h   = (_Float16*)alloc((size_t)G_ * H_ * 2);
    _Float16* w_ih1_h   = (_Float16*)alloc((size_t)G_ * H_ * 2);
    _Float16* w_hh1_h   = (_Float16*)alloc((size_t)G_ * H_ * 2);
    _Float16* h0a_h     = (_Float16*)alloc((size_t)B_ * H_ * 2);
    _Float16* h0b_h     = (_Float16*)alloc((size_t)B_ * H_ * 2);
    _Float16* h1_h      = (_Float16*)alloc((size_t)B_ * H_ * 2);
    _Float16* comb_in_h = (_Float16*)alloc((size_t)B_ * CK_ * 2);
    _Float16* comb_h    = (_Float16*)alloc((size_t)B_ * IN_ * 2);
    float*    x_rnn_f   = (float*)   alloc((size_t)B_ * IN_ * 4);
    float*    logits_f  = (float*)   alloc((size_t)B_ * T_ * 4);
    float*    gates_f   = (float*)   alloc((size_t)B_ * G_ * 4);

    auto conv = [&](const float* s, _Float16* d, int n) {
        f32_to_f16_kernel<<<(n + 255) / 256, 256, 0, stream>>>(s, d, n);
    };

    // 0) fp32 -> f16 repacks (tiny vs. the 2 GB stream)
    conv(attn_w, attn_w_h, T_ * AK_);
    conv(comb_w, comb_w_h, IN_ * CK_);
    conv(w_ih0,  w_ih0_h,  G_ * IN_);
    conv(w_hh0,  w_hh0_h,  G_ * H_);
    conv(w_ih1,  w_ih1_h,  G_ * H_);
    conv(w_hh1,  w_hh1_h,  G_ * H_);
    conv(h0,            h0a_h, B_ * H_);
    conv(h0 + (size_t)B_ * H_, h0b_h, B_ * H_);

    // 1) embeddings + concat rows
    build_inputs_kernel<<<B_, 256, 0, stream>>>(x, x_emb, h0, c0,
                                                emb0, emb1, emb2, emb3,
                                                x_rnn_f, attn_in_h, comb_in_h);

    // 2) attention logits: (B x T) = attn_in @ attn_w^T + attn_b   [WMMA]
    wmma_gemm_dual<<<(B_ / 16) * (T_ / 16) / 8, 256, 0, stream>>>(
        attn_in_h, attn_w_h, AK_, nullptr, nullptr, 0,
        attn_b, nullptr, logits_f, nullptr, B_, T_);

    // 3) softmax -> aw (written straight into d_out)
    softmax_kernel<<<B_, 256, 0, stream>>>(logits_f, out + OFF_AW);

    // 4) ctx einsum (bandwidth floor: 2 GB of encoder_outputs)
    ctx_kernel<<<B_, 256, 0, stream>>>(out + OFF_AW, enc, comb_in_h);

    // 5) comb: (B x IN) = [x_rnn|ctx] @ comb_w^T + comb_b   [WMMA]
    wmma_gemm_dual<<<(B_ / 16) * (IN_ / 16) / 8, 256, 0, stream>>>(
        comb_in_h, comb_w_h, CK_, nullptr, nullptr, 0,
        comb_b, nullptr, nullptr, comb_h, B_, IN_);

    // 6) gates0 = comb @ w_ih0^T + h0[0] @ w_hh0^T + biases   [WMMA dual-K]
    wmma_gemm_dual<<<(B_ / 16) * (G_ / 16) / 8, 256, 0, stream>>>(
        comb_h, w_ih0_h, IN_, h0a_h, w_hh0_h, H_,
        b_ih0, b_hh0, gates_f, nullptr, B_, G_);

    // 7) LSTM0 elementwise -> h1, c1
    lstm_elem_kernel<<<(B_ * H_) / 256, 256, 0, stream>>>(
        gates_f, c0, out + OFF_H1, out + OFF_C1, h1_h);

    // 8) gates1 = h1 @ w_ih1^T + h0[1] @ w_hh1^T + biases   [WMMA dual-K]
    wmma_gemm_dual<<<(B_ / 16) * (G_ / 16) / 8, 256, 0, stream>>>(
        h1_h, w_ih1_h, H_, h0b_h, w_hh1_h, H_,
        b_ih1, b_hh1, gates_f, nullptr, B_, G_);

    // 9) LSTM1 elementwise -> h2, c2
    lstm_elem_kernel<<<(B_ * H_) / 256, 256, 0, stream>>>(
        gates_f, c0 + (size_t)B_ * H_, out + OFF_H2, out + OFF_C2, nullptr);

    // 10) prediction head
    pred_kernel<<<B_, 256, 0, stream>>>(out + OFF_H2, pred_w, pred_b, out);
}
